// WayfinderAttentionMLX_66821101191647
// MI455X (gfx1250) — matrix operand order repaired and checked by reading, not compiled, not run on hardware
//
#include <hip/hip_runtime.h>

// CDNA5 / gfx1250, wave32. One query (h,t) per wave; WMMA f16 with f32 accum.

typedef _Float16 v16h __attribute__((ext_vector_type(16)));
typedef float    v8f  __attribute__((ext_vector_type(8)));

#define NH_  8
#define T_   2048
#define DH_  64
#define ND_  64
#define WPB  8          // waves per block

__device__ __forceinline__ void pack8(v16h& a, int base, float4 f0, float4 f1) {
  a[base + 0] = (_Float16)f0.x; a[base + 1] = (_Float16)f0.y;
  a[base + 2] = (_Float16)f0.z; a[base + 3] = (_Float16)f0.w;
  a[base + 4] = (_Float16)f1.x; a[base + 5] = (_Float16)f1.y;
  a[base + 6] = (_Float16)f1.z; a[base + 7] = (_Float16)f1.w;
}

// pack with scale folded in (used for q * 1/sqrt(DH); 0.125 is a power of two,
// so scaling before the f16 convert is bit-exact vs scaling the f32 score)
__device__ __forceinline__ void pack8s(v16h& a, int base, float4 f0, float4 f1, float s) {
  a[base + 0] = (_Float16)(f0.x * s); a[base + 1] = (_Float16)(f0.y * s);
  a[base + 2] = (_Float16)(f0.z * s); a[base + 3] = (_Float16)(f0.w * s);
  a[base + 4] = (_Float16)(f1.x * s); a[base + 5] = (_Float16)(f1.y * s);
  a[base + 6] = (_Float16)(f1.z * s); a[base + 7] = (_Float16)(f1.w * s);
}

__global__ __launch_bounds__(256, 1) void wayfinder_attn_wmma(
    const float* __restrict__ q, const float* __restrict__ k,
    const float* __restrict__ v, const float* __restrict__ etb,
    const int* __restrict__ neigh, const int* __restrict__ etype,
    float* __restrict__ out)
{
  __shared__ int   s_sidx[WPB][ND_];
  __shared__ float s_badd[WPB][ND_];

  const int lane  = threadIdx.x & 31;
  const int wslot = (int)(threadIdx.x >> 5);
  const int qid   = blockIdx.x * WPB + wslot;   // flat h*T + t, in [0, 16384)
  const int h     = qid >> 11;                  // / 2048
  const int t     = qid & (T_ - 1);

  const int half16 = lane >> 4;                 // 0 for lanes 0-15, 1 for 16-31
  const int lm     = lane & 15;
  const int c0     = half16 * 8;                // A-layout channel-chunk base

  // ---- stage clipped gather indices + additive (bias | -inf mask) into LDS ----
  {
    const int  n0 = lane * 2;
    const long ib = (long)qid * ND_ + n0;
    int2 ni = *(const int2*)(neigh + ib);
    int2 et = *(const int2*)(etype + ib);
    float b0 = (et.x == 0) ? 0.0f : etb[et.x - 1];   // full_bias = [0, etb...]
    float b1 = (et.y == 0) ? 0.0f : etb[et.y - 1];
    bool ok0 = (ni.x >= 0) && (ni.x <= t);           // causal neighbor mask
    bool ok1 = (ni.y >= 0) && (ni.y <= t);
    s_sidx[wslot][n0]     = min(max(ni.x, 0), T_ - 1);
    s_sidx[wslot][n0 + 1] = min(max(ni.y, 0), T_ - 1);
    s_badd[wslot][n0]     = ok0 ? b0 : -1e30f;
    s_badd[wslot][n0 + 1] = ok1 ? b1 : -1e30f;
  }
  __syncthreads();

  // ---- B operand for scores: (q / sqrt(DH)) replicated across all 16 columns ----
  // B(32x16) lane layout: N = lane%16, K = c0+{0..7} then 16+c0+{0..7}, + 32*kt
  const float* qrow = q + (long)qid * DH_;
  v16h bq[2];
  #pragma unroll
  for (int kt = 0; kt < 2; ++kt) {
    const float* p = qrow + kt * 32 + half16 * 8;
    pack8s(bq[kt], 0, *(const float4*)(p),     *(const float4*)(p + 4),  0.125f);
    pack8s(bq[kt], 8, *(const float4*)(p + 16), *(const float4*)(p + 20), 0.125f);
  }

  // ---- scores S[64] = K_gathered(64x64) * q/8 : 4 M-tiles x 2 K-tiles ----
  // A(16x32) lane layout: M = lane%16; channels c0+{0..7} then 16+c0+{0..7}
  const float* kbase = k + (long)h * (T_ * DH_);
  v8f acc[4];
  #pragma unroll
  for (int mt = 0; mt < 4; ++mt) {
    const int    row = s_sidx[wslot][mt * 16 + lm];
    const float* kr  = kbase + (long)row * DH_;
    v8f a8 = {};
    #pragma unroll
    for (int kt = 0; kt < 2; ++kt) {
      v16h a;
      const float* p = kr + kt * 32 + c0;
      pack8(a, 0, *(const float4*)(p),      *(const float4*)(p + 4));
      pack8(a, 8, *(const float4*)(p + 16), *(const float4*)(p + 20));
      a8 = __builtin_amdgcn_wmma_f32_16x16x32_f16(false, a, false, bq[kt],
                                                  (short)0, a8, false, false);
    }
    acc[mt] = a8;
  }

  // ---- masked softmax over 64 neighbors (f32, matches reference semantics) ----
  // acc[mt][r] = score of neighbor (mt*16 + c0 + r); lane halves own disjoint sets.
  float sc[4][8];
  float mx = -3.4e38f;
  #pragma unroll
  for (int mt = 0; mt < 4; ++mt) {
    float4 ba = *(const float4*)&s_badd[wslot][mt * 16 + c0];
    float4 bb = *(const float4*)&s_badd[wslot][mt * 16 + c0 + 4];
    const float bl[8] = {ba.x, ba.y, ba.z, ba.w, bb.x, bb.y, bb.z, bb.w};
    #pragma unroll
    for (int r = 0; r < 8; ++r) {
      float s = acc[mt][r] + bl[r];
      sc[mt][r] = s;
      mx = fmaxf(mx, s);
    }
  }
  mx = fmaxf(mx, __shfl_xor(mx, 16, 32));

  float sum = 0.0f;
  #pragma unroll
  for (int mt = 0; mt < 4; ++mt)
    #pragma unroll
    for (int r = 0; r < 8; ++r) {
      float e = (sc[mt][r] > -1e29f) ? __expf(sc[mt][r] - mx) : 0.0f;
      sc[mt][r] = e;
      sum += e;
    }
  sum += __shfl_xor(sum, 16, 32);
  const float inv = 1.0f / fmaxf(sum, 1e-9f);

  // ---- A operand for output: w replicated rows. The score-accumulator layout
  // lands exactly on the A K-chunk pattern: aw[kt][i] = w[mt=2kt+(i>=8), r=i%8].
  v16h aw[2];
  #pragma unroll
  for (int kt = 0; kt < 2; ++kt)
    #pragma unroll
    for (int i = 0; i < 16; ++i)
      aw[kt][i] = (_Float16)(sc[2 * kt + (i >> 3)][i & 7] * inv);

  // ---- y[64] = w(1x64) * V_gathered(64x64) : 2 K-tiles x 4 column-tiles ----
  // kt-outer: hoist 16 gathered-row bases once per K-tile; the 4 column tiles
  // then load at immediate offsets 0/64/128/192B from the same base address.
  // Across the 16 lanes of a half each load covers a contiguous 64B row run.
  const float* vbase = v + (long)h * (T_ * DH_);
  v8f y0 = {}, y1 = {}, y2 = {}, y3 = {};
  #pragma unroll
  for (int kt = 0; kt < 2; ++kt) {
    const int nb = kt * 32 + half16 * 8;       // B layout: K rows c0+{0..7}, 16+c0+{0..7}
    int4 r0 = *(const int4*)&s_sidx[wslot][nb + 0];
    int4 r1 = *(const int4*)&s_sidx[wslot][nb + 4];
    int4 r2 = *(const int4*)&s_sidx[wslot][nb + 16];
    int4 r3 = *(const int4*)&s_sidx[wslot][nb + 20];
    const int rows[16] = {r0.x, r0.y, r0.z, r0.w, r1.x, r1.y, r1.z, r1.w,
                          r2.x, r2.y, r2.z, r2.w, r3.x, r3.y, r3.z, r3.w};
    v16h b0, b1, b2, b3;
    #pragma unroll
    for (int i = 0; i < 16; ++i) {
      const float* rp = vbase + (long)rows[i] * DH_ + lm;
      b0[i] = (_Float16)rp[0];                 // channel ct*16 + lm, ct = 0..3
      b1[i] = (_Float16)rp[16];
      b2[i] = (_Float16)rp[32];
      b3[i] = (_Float16)rp[48];
    }
    y0 = __builtin_amdgcn_wmma_f32_16x16x32_f16(false, aw[kt], false, b0, (short)0, y0, false, false);
    y1 = __builtin_amdgcn_wmma_f32_16x16x32_f16(false, aw[kt], false, b1, (short)0, y1, false, false);
    y2 = __builtin_amdgcn_wmma_f32_16x16x32_f16(false, aw[kt], false, b2, (short)0, y2, false, false);
    y3 = __builtin_amdgcn_wmma_f32_16x16x32_f16(false, aw[kt], false, b3, (short)0, y3, false, false);
  }

  float* orow = out + (long)qid * DH_;
  if (lane < 16) {                              // D row 0, N = lane
    orow[lm]      = y0[0];
    orow[16 + lm] = y1[0];
    orow[32 + lm] = y2[0];
    orow[48 + lm] = y3[0];
  }
}

extern "C" void kernel_launch(void* const* d_in, const int* in_sizes, int n_in,
                              void* d_out, int out_size, void* d_ws, size_t ws_size,
                              hipStream_t stream) {
  (void)in_sizes; (void)n_in; (void)out_size; (void)d_ws; (void)ws_size;
  const float* q     = (const float*)d_in[0];
  const float* k     = (const float*)d_in[1];
  const float* v     = (const float*)d_in[2];
  const float* etb   = (const float*)d_in[3];
  const int*   neigh = (const int*)d_in[4];
  const int*   etype = (const int*)d_in[5];
  float*       out   = (float*)d_out;

  dim3 grid((NH_ * T_) / WPB);   // 2048 blocks
  dim3 block(32 * WPB);          // 256 threads = 8 wave32
  hipLaunchKernelGGL(wayfinder_attn_wmma, grid, block, 0, stream,
                     q, k, v, etb, neigh, etype, out);
}
